// CausalMAC_55903294324922
// MI455X (gfx1250) — matrix operand order repaired and checked by reading, not compiled, not run on hardware
//
#include <hip/hip_runtime.h>

// ---------------------------------------------------------------------------
// CDNA5 (gfx1250) WMMA implementation of CausalMAC reference.
// bf16 WMMA 16x16x32 with f32 accumulation; fp32 master activations for
// layernorm/softmax; relation MLP decomposed: pair@W1 = x_i@W1a + x_j@W1b.
// All WMMA B-operands are stored transposed so fragment loads are contiguous
// 32-byte (2x global_load_b128) reads. The generic GEMM register-blocks
// 32x64 per wave (2x4 tiles, 8 accumulators) for A/B fragment reuse.
// ---------------------------------------------------------------------------

typedef __attribute__((ext_vector_type(16))) __bf16 v16bf;
typedef __attribute__((ext_vector_type(8)))  __bf16 v8bf;
typedef __attribute__((ext_vector_type(8)))  float  v8f;

#define B_  16
#define N_  256
#define F_  64
#define D_  128
#define H_  4
#define DH_ 32
#define L_  2
#define DFF_ 512
#define ROWS_ (B_ * N_)          // 4096
#define P_  (N_ * (N_ - 1))      // 65280

// ----------------------- WMMA fragment loaders -----------------------------
// A fragment, 16x32 bf16, source row-major with leading dim lda.
// lane<16: M=lane, K = 0..15; lane>=16: M=lane-16, K = 16..31
__device__ __forceinline__ v16bf fragA_rm(const __bf16* A, int lda, int lane) {
  int m = lane & 15, kb = (lane >> 4) * 16;
  const __bf16* p = A + m * lda + kb;
  v8bf lo = *(const v8bf*)(p);
  v8bf hi = *(const v8bf*)(p + 8);
  return __builtin_shufflevector(lo, hi, 0, 1, 2, 3, 4, 5, 6, 7,
                                 8, 9, 10, 11, 12, 13, 14, 15);
}

// B fragment, 32x16 bf16, where B is stored TRANSPOSED (NxK row-major):
// B[k][n] == Bt[n*ldt + k]. Contiguous 16-element (32B) read per lane.
// lane<16: N=lane, K=0..15 ; lane>=16: N=lane-16, K=16..31
__device__ __forceinline__ v16bf fragB_t(const __bf16* Bt, int ldt, int lane) {
  int n = lane & 15, kb = (lane >> 4) * 16;
  const __bf16* p = Bt + n * ldt + kb;
  v8bf lo = *(const v8bf*)(p);
  v8bf hi = *(const v8bf*)(p + 8);
  return __builtin_shufflevector(lo, hi, 0, 1, 2, 3, 4, 5, 6, 7,
                                 8, 9, 10, 11, 12, 13, 14, 15);
}

// ----------------------- generic GEMM (wave per 32x64 block) ---------------
// A: MxK row-major bf16. Bt: weight stored transposed, Bt[n*ldb + k].
// Each wave: 2 M-tiles x 4 N-tiles, 8 f32 accumulators, fragments reused.
__global__ void gemm_bf16(const __bf16* __restrict__ A, const __bf16* __restrict__ Bt,
                          int ldb, const float* __restrict__ bias, float* C, __bf16* Cbf,
                          int M, int N, int K, int relu) {
  int lane = threadIdx.x & 31;
  int wtile = blockIdx.x * (blockDim.x >> 5) + (threadIdx.x >> 5);
  int wN = N >> 6;                       // 64-wide blocks
  if (wtile >= (M >> 5) * wN) return;    // 32-high blocks
  int tm = wtile / wN, tn = wtile % wN;

  const __bf16* Ab = A + (long)(tm * 32) * K;
  const __bf16* Bb = Bt + (long)(tn * 64) * ldb;
  v8f acc[2][4] = {};
  for (int k0 = 0; k0 < K; k0 += 32) {
    v16bf a[2], b[4];
#pragma unroll
    for (int i = 0; i < 2; ++i) a[i] = fragA_rm(Ab + (long)(i * 16) * K + k0, K, lane);
#pragma unroll
    for (int j = 0; j < 4; ++j) b[j] = fragB_t(Bb + (long)(j * 16) * ldb + k0, ldb, lane);
#pragma unroll
    for (int i = 0; i < 2; ++i)
#pragma unroll
      for (int j = 0; j < 4; ++j)
        acc[i][j] = __builtin_amdgcn_wmma_f32_16x16x32_bf16(
            false, a[i], false, b[j], (short)0, acc[i][j], false, false);
  }
  int n = lane & 15, mb = (lane >> 4) * 8;
#pragma unroll
  for (int i = 0; i < 2; ++i) {
#pragma unroll
    for (int j = 0; j < 4; ++j) {
      int col = tn * 64 + j * 16 + n;
      float bv = bias ? bias[col] : 0.0f;
#pragma unroll
      for (int r = 0; r < 8; ++r) {
        float v = acc[i][j][r] + bv;
        if (relu) v = fmaxf(v, 0.0f);
        long idx = (long)(tm * 32 + i * 16 + mb + r) * N + col;
        if (C)   C[idx] = v;
        if (Cbf) Cbf[idx] = (__bf16)v;
      }
    }
  }
}

// ----------------------- attention: scores = Q K^T / sqrt(DH) --------------
// qkv_bf: [4096, 384] bf16 (q | k | v, head-contiguous 32).  K^T fragments are
// contiguous reads along the head dim of token rows.
__global__ void attn_scores(const __bf16* __restrict__ qkv, float* __restrict__ scores) {
  int lane = threadIdx.x & 31;
  int tile = blockIdx.x * (blockDim.x >> 5) + (threadIdx.x >> 5); // 16384 tiles
  int tn = tile & 15, tm = (tile >> 4) & 15, bh = tile >> 8;      // bh = b*H+h
  int b = bh >> 2, h = bh & 3;

  const __bf16* qb = qkv + (long)(b * N_) * (3 * D_) + h * DH_;
  const __bf16* kb = qkv + (long)(b * N_) * (3 * D_) + D_ + h * DH_;
  v16bf a  = fragA_rm(qb + (long)(tm * 16) * (3 * D_), 3 * D_, lane);
  v16bf bt = fragB_t(kb + (long)(tn * 16) * (3 * D_), 3 * D_, lane);
  v8f acc = {};
  acc = __builtin_amdgcn_wmma_f32_16x16x32_bf16(false, a, false, bt, (short)0, acc,
                                                false, false);
  const float sc = 0.17677669529663687f; // 1/sqrt(32)
  int n = lane & 15, mb = (lane >> 4) * 8;
  float* out = scores + ((long)bh * N_ + tm * 16) * N_ + tn * 16;
#pragma unroll
  for (int r = 0; r < 8; ++r) out[(mb + r) * N_ + n] = acc[r] * sc;
}

// ----------------------- softmax over 256, wave per row --------------------
__global__ void softmax_rows(const float* __restrict__ scores, __bf16* __restrict__ attn) {
  int lane = threadIdx.x & 31;
  int row = blockIdx.x * (blockDim.x >> 5) + (threadIdx.x >> 5); // 16384 rows
  const float* s = scores + (long)row * N_;
  float v[8], mx = -1e30f;
#pragma unroll
  for (int e = 0; e < 8; ++e) { v[e] = s[lane + 32 * e]; mx = fmaxf(mx, v[e]); }
  for (int off = 16; off; off >>= 1) mx = fmaxf(mx, __shfl_xor(mx, off, 32));
  float sum = 0.0f;
#pragma unroll
  for (int e = 0; e < 8; ++e) { v[e] = __expf(v[e] - mx); sum += v[e]; }
  for (int off = 16; off; off >>= 1) sum += __shfl_xor(sum, off, 32);
  float inv = 1.0f / sum;
  __bf16* o = attn + (long)row * N_;
#pragma unroll
  for (int e = 0; e < 8; ++e) o[lane + 32 * e] = (__bf16)(v[e] * inv);
}

// ----------------------- V transpose: vt[bh][dh][tok] ----------------------
__global__ void vt_kernel(const __bf16* __restrict__ qkv, __bf16* __restrict__ vt) {
  int idx = blockIdx.x * blockDim.x + threadIdx.x; // B*H*DH*N = 524288
  int n = idx & 255, d = (idx >> 8) & 31, bh = idx >> 13;
  int b = bh >> 2, h = bh & 3;
  vt[((long)bh * DH_ + d) * N_ + n] =
      qkv[(long)(b * N_ + n) * (3 * D_) + 2 * D_ + h * DH_ + d];
}

// ----------------------- attention: O = P V (K=256) ------------------------
__global__ void attn_av(const __bf16* __restrict__ attn, const __bf16* __restrict__ vt,
                        __bf16* __restrict__ o_bf) {
  int lane = threadIdx.x & 31;
  int tile = blockIdx.x * (blockDim.x >> 5) + (threadIdx.x >> 5); // 2048 tiles
  int tn = tile & 1, tm = (tile >> 1) & 15, bh = tile >> 5;
  int b = bh >> 2, h = bh & 3;

  const __bf16* Ab = attn + (long)bh * N_ * N_ + (long)(tm * 16) * N_;
  const __bf16* Vt = vt + (long)bh * DH_ * N_ + (long)(tn * 16) * N_; // [dh][tok]
  v8f acc = {};
  for (int k0 = 0; k0 < N_; k0 += 32) {
    v16bf a = fragA_rm(Ab + k0, N_, lane);
    v16bf v = fragB_t(Vt + k0, N_, lane);
    acc = __builtin_amdgcn_wmma_f32_16x16x32_bf16(false, a, false, v, (short)0, acc,
                                                  false, false);
  }
  int n = lane & 15, mb = (lane >> 4) * 8;
  long base = ((long)b * N_ + tm * 16) * D_ + h * DH_ + tn * 16;
#pragma unroll
  for (int r = 0; r < 8; ++r) o_bf[base + (long)(mb + r) * D_ + n] = (__bf16)acc[r];
}

// ----------------------- LayerNorm (+optional residual), wave per row ------
__global__ void ln_kernel(const float* in, const float* resid,
                          const float* __restrict__ g, const float* __restrict__ beta,
                          float* out_f, __bf16* out_bf) {
  int lane = threadIdx.x & 31;
  int row = blockIdx.x * (blockDim.x >> 5) + (threadIdx.x >> 5); // 4096 rows
  long base = (long)row * D_;
  float v[4], s = 0.0f;
#pragma unroll
  for (int e = 0; e < 4; ++e) {
    int d = lane + 32 * e;
    float x = in[base + d];
    if (resid) x += resid[base + d];
    v[e] = x; s += x;
  }
  for (int off = 16; off; off >>= 1) s += __shfl_xor(s, off, 32);
  float mu = s * (1.0f / D_), var = 0.0f;
#pragma unroll
  for (int e = 0; e < 4; ++e) { float d0 = v[e] - mu; var += d0 * d0; }
  for (int off = 16; off; off >>= 1) var += __shfl_xor(var, off, 32);
  float inv = rsqrtf(var * (1.0f / D_) + 1e-5f);
#pragma unroll
  for (int e = 0; e < 4; ++e) {
    int d = lane + 32 * e;
    float y = (v[e] - mu) * inv * g[d] + beta[d];
    out_f[base + d] = y;
    out_bf[base + d] = (__bf16)y;
  }
}

// ----------------------- relation pair head --------------------------------
// out[b, p(i,j)] = sum_d relu(A[b,i,d] + Bm[b,j,d] + b1[d]) * w2[d] + b2
__global__ void relpair(const float* __restrict__ Aacc, const float* __restrict__ Bacc,
                        const float* __restrict__ b1, const float* __restrict__ w2,
                        const float* __restrict__ b2, float* __restrict__ out) {
  __shared__ float sA[D_], sW[D_];
  int bi = blockIdx.x, b = bi >> 8, i = bi & 255;
  int t = threadIdx.x; // 256 threads, one j per thread
  if (t < D_) {
    sA[t] = Aacc[((long)b * N_ + i) * D_ + t] + b1[t];
    sW[t] = w2[t];
  }
  __syncthreads();
  int j = t;
  if (j == i) return;
  const float* Br = Bacc + ((long)b * N_ + j) * D_;
  float s = 0.0f;
#pragma unroll 4
  for (int d = 0; d < D_; ++d) s = fmaf(fmaxf(sA[d] + Br[d], 0.0f), sW[d], s);
  int p = i * (N_ - 1) + (j < i ? j : j - 1);
  out[(long)b * P_ + p] = s + b2[0];
}

// ----------------------- fp32 -> bf16 conversions --------------------------
__global__ void f2bf(const float* __restrict__ in, __bf16* __restrict__ out, int n) {
  int i = blockIdx.x * blockDim.x + threadIdx.x;
  if (i < n) out[i] = (__bf16)in[i];
}

// transpose-convert: in is KxN f32 (row-major), out is NxK bf16 (row-major)
__global__ void f2bf_t(const float* __restrict__ in, __bf16* __restrict__ out,
                       int K, int N) {
  int idx = blockIdx.x * blockDim.x + threadIdx.x;
  if (idx >= K * N) return;
  int n = idx % N, k = idx / N;      // reads coalesced over n
  out[(long)n * K + k] = (__bf16)in[idx];
}

// ===========================================================================
extern "C" void kernel_launch(void* const* d_in, const int* in_sizes, int n_in,
                              void* d_out, int out_size, void* d_ws, size_t ws_size,
                              hipStream_t stream) {
  const float* agents   = (const float*)d_in[0];
  const float* enc_w    = (const float*)d_in[1];
  const float* enc_b    = (const float*)d_in[2];
  const float* enc_g    = (const float*)d_in[3];
  const float* enc_beta = (const float*)d_in[4];
  const float* qkv_w    = (const float*)d_in[5];
  const float* qkv_b    = (const float*)d_in[6];
  const float* attn_ow  = (const float*)d_in[7];
  const float* attn_ob  = (const float*)d_in[8];
  const float* ln1_g    = (const float*)d_in[9];
  const float* ln1_b    = (const float*)d_in[10];
  const float* ffn_w1   = (const float*)d_in[11];
  const float* ffn_b1   = (const float*)d_in[12];
  const float* ffn_w2   = (const float*)d_in[13];
  const float* ffn_b2   = (const float*)d_in[14];
  const float* ln2_g    = (const float*)d_in[15];
  const float* ln2_b    = (const float*)d_in[16];
  const float* rel_w1   = (const float*)d_in[17];
  const float* rel_b1   = (const float*)d_in[18];
  const float* rel_w2   = (const float*)d_in[19];
  const float* rel_b2   = (const float*)d_in[20];
  float* out = (float*)d_out;

  // ---- workspace arena -----------------------------------------------------
  char* w = (char*)d_ws;
  auto arena = [&](size_t bytes) { char* p = w; w += (bytes + 255) & ~(size_t)255; return p; };

  __bf16* agents_bf = (__bf16*)arena((size_t)ROWS_ * F_ * 2);
  __bf16* wt_enc  = (__bf16*)arena((size_t)D_ * F_ * 2);            // [128][64]
  __bf16* wt_qkv  = (__bf16*)arena((size_t)L_ * 3 * D_ * D_ * 2);   // [384][128] x L
  __bf16* wt_ow   = (__bf16*)arena((size_t)L_ * D_ * D_ * 2);       // [128][128] x L
  __bf16* wt_w1   = (__bf16*)arena((size_t)L_ * DFF_ * D_ * 2);     // [512][128] x L
  __bf16* wt_w2   = (__bf16*)arena((size_t)L_ * D_ * DFF_ * 2);     // [128][512] x L
  __bf16* wt_rel1 = (__bf16*)arena((size_t)D_ * 2 * D_ * 2);        // [128][256]
  float*  x_f     = (float*)arena((size_t)ROWS_ * D_ * 4);
  __bf16* x_bf    = (__bf16*)arena((size_t)ROWS_ * D_ * 2);
  float*  buf_f   = (float*)arena((size_t)ROWS_ * DFF_ * 4);
  __bf16* buf_bf  = (__bf16*)arena((size_t)ROWS_ * DFF_ * 2);
  __bf16* qkv_bf  = (__bf16*)arena((size_t)ROWS_ * 3 * D_ * 2);
  float*  sc_f    = (float*)arena((size_t)B_ * H_ * N_ * N_ * 4);
  __bf16* attn_bf = (__bf16*)arena((size_t)B_ * H_ * N_ * N_ * 2);
  __bf16* vt_bf   = (__bf16*)arena((size_t)B_ * H_ * DH_ * N_ * 2);
  __bf16* o_bf    = (__bf16*)arena((size_t)ROWS_ * D_ * 2);
  float*  Aacc    = (float*)arena((size_t)ROWS_ * D_ * 4);
  float*  Bacc    = (float*)arena((size_t)ROWS_ * D_ * 4);
  (void)ws_size; (void)n_in; (void)in_sizes; (void)out_size;

  auto conv = [&](const float* src, __bf16* dst, int n) {
    f2bf<<<(n + 255) / 256, 256, 0, stream>>>(src, dst, n);
  };
  auto conv_t = [&](const float* src, __bf16* dst, int K, int N) {
    f2bf_t<<<(K * N + 255) / 256, 256, 0, stream>>>(src, dst, K, N);
  };
  auto gemm = [&](const __bf16* A, const __bf16* Bt, int ldb, const float* bias,
                  float* C, __bf16* Cbf, int M, int N, int K, int relu) {
    int tiles = (M >> 5) * (N >> 6);   // 32x64 blocks per wave
    gemm_bf16<<<(tiles + 7) / 8, 256, 0, stream>>>(A, Bt, ldb, bias, C, Cbf,
                                                   M, N, K, relu);
  };

  // ---- weight & input conversion (weights stored transposed) ---------------
  conv(agents, agents_bf, ROWS_ * F_);
  conv_t(enc_w, wt_enc, F_, D_);
  for (int l = 0; l < L_; ++l) {
    conv_t(qkv_w  + (size_t)l * D_ * 3 * D_, wt_qkv + (size_t)l * 3 * D_ * D_, D_, 3 * D_);
    conv_t(attn_ow + (size_t)l * D_ * D_,    wt_ow  + (size_t)l * D_ * D_,     D_, D_);
    conv_t(ffn_w1 + (size_t)l * D_ * DFF_,   wt_w1  + (size_t)l * DFF_ * D_,   D_, DFF_);
    conv_t(ffn_w2 + (size_t)l * DFF_ * D_,   wt_w2  + (size_t)l * D_ * DFF_,   DFF_, D_);
  }
  conv_t(rel_w1, wt_rel1, 2 * D_, D_);  // [128][256]: col k -> rel_w1 row k

  // ---- agent encoder: relu(agents @ enc_w + b) then LN --------------------
  gemm(agents_bf, wt_enc, F_, enc_b, buf_f, nullptr, ROWS_, D_, F_, 1);
  ln_kernel<<<ROWS_ / 8, 256, 0, stream>>>(buf_f, nullptr, enc_g, enc_beta, x_f, x_bf);

  // ---- transformer layers --------------------------------------------------
  for (int l = 0; l < L_; ++l) {
    gemm(x_bf, wt_qkv + (size_t)l * 3 * D_ * D_, D_, qkv_b + l * 3 * D_,
         nullptr, qkv_bf, ROWS_, 3 * D_, D_, 0);
    attn_scores<<<(B_ * H_ * 16 * 16) / 8, 256, 0, stream>>>(qkv_bf, sc_f);
    softmax_rows<<<(B_ * H_ * N_) / 8, 256, 0, stream>>>(sc_f, attn_bf);
    vt_kernel<<<(B_ * H_ * DH_ * N_) / 256, 256, 0, stream>>>(qkv_bf, vt_bf);
    attn_av<<<(B_ * H_ * 16 * 2) / 8, 256, 0, stream>>>(attn_bf, vt_bf, o_bf);
    gemm(o_bf, wt_ow + (size_t)l * D_ * D_, D_, attn_ob + l * D_,
         buf_f, nullptr, ROWS_, D_, D_, 0);
    ln_kernel<<<ROWS_ / 8, 256, 0, stream>>>(x_f, buf_f, ln1_g + l * D_, ln1_b + l * D_,
                                             x_f, x_bf);
    gemm(x_bf, wt_w1 + (size_t)l * DFF_ * D_, D_, ffn_b1 + l * DFF_,
         nullptr, buf_bf, ROWS_, DFF_, D_, 1);
    gemm(buf_bf, wt_w2 + (size_t)l * D_ * DFF_, DFF_, ffn_b2 + l * D_,
         buf_f, nullptr, ROWS_, D_, DFF_, 0);
    ln_kernel<<<ROWS_ / 8, 256, 0, stream>>>(x_f, buf_f, ln2_g + l * D_, ln2_b + l * D_,
                                             x_f, x_bf);
  }

  // ---- relation head: decomposed pair MLP ---------------------------------
  // wt_rel1 is [128][256]: row n holds rel_w1[:,n]; W1a^T = cols 0..127 (ld 256),
  // W1b^T = cols 128..255 (ld 256).
  gemm(x_bf, wt_rel1,       2 * D_, nullptr, Aacc, nullptr, ROWS_, D_, D_, 0);
  gemm(x_bf, wt_rel1 + D_,  2 * D_, nullptr, Bacc, nullptr, ROWS_, D_, D_, 0);
  relpair<<<B_ * N_, 256, 0, stream>>>(Aacc, Bacc, rel_b1, rel_w2, rel_b2, out);
}
